// GCNEdge2Cluster_modularity_51144470561271
// MI455X (gfx1250) — matrix operand order repaired and checked by reading, not compiled, not run on hardware
//
#include <hip/hip_runtime.h>
#include <stdint.h>
#include <math.h>

#define DEVINL __device__ __forceinline__

typedef __attribute__((ext_vector_type(16))) __bf16 v16bf;
typedef __attribute__((ext_vector_type(8)))  float  v8f;

constexpr int   N    = 100000;
constexpr int   E    = 3200000;
constexpr int   F_IN = 128;
constexpr int   H    = 256;
constexpr int   K    = 30;
constexpr int   KP   = 32;     // K padded to 2 WMMA column tiles
constexpr float REGC = 0.01f;

// ---------------- helpers ----------------

DEVINL unsigned short f2bf(float f) {            // round-to-nearest-even f32 -> bf16
  uint32_t u = __builtin_bit_cast(uint32_t, f);
  u = u + 0x7FFFu + ((u >> 16) & 1u);
  return (unsigned short)(u >> 16);
}

DEVINL void atomAddF(float* p, float v) {        // native global_atomic_add_f32
  unsafeAtomicAdd(p, v);
}

union Frag { v16bf v; uint32_t u[8]; };

// A fragment: 16x32 bf16, rows r0..r0+15, K = k0..k0+31.
// ISA layout: lanes 0-15 -> M=lane, K 0-7 & 16-23 ; lanes 16-31 -> M=lane-16, K 8-15 & 24-31.
// K pairs (2j,2j+1) are contiguous in row-major memory -> dword loads.
DEVINL v16bf load_a(const unsigned short* __restrict__ X, int ld, int r0, int k0) {
  const int l  = threadIdx.x & 31;
  const int m  = l & 15;
  const int kh = (l >> 4) << 3;                  // 0 or 8
  Frag f;
  const uint32_t* base = (const uint32_t*)(X + (size_t)(r0 + m) * ld + k0);
#pragma unroll
  for (int j = 0; j < 8; ++j) {
    const int k = (j < 4) ? (kh + 2 * j) : (16 + kh + 2 * (j - 4));
    f.u[j] = base[k >> 1];
  }
  return f.v;
}

// B fragment: 32x16 bf16 from PRE-TRANSPOSED weights Wt[n][k] (ldk = K-dim).
// Layout: lanes 0-15 -> N=lane, K 0-7 & 16-23 ; lanes 16-31 -> N=lane-16, K 8-15 & 24-31 (matches A).
DEVINL v16bf load_b(const unsigned short* __restrict__ Wt, int ldk, int k0, int c0) {
  const int l  = threadIdx.x & 31;
  const int n  = l & 15;
  const int kh = (l >> 4) << 3;                  // 0 or 8
  Frag f;
  const uint32_t* base = (const uint32_t*)(Wt + (size_t)(c0 + n) * ldk + k0);
#pragma unroll
  for (int j = 0; j < 8; ++j) f.u[j] = base[((j < 4) ? (kh + 2 * j) : (16 + kh + 2 * (j - 4))) >> 1];
  return f.v;
}

DEVINL v8f wmma_bf16(v16bf a, v16bf b, v8f c) {
  return __builtin_amdgcn_wmma_f32_16x16x32_bf16(false, a, false, b, (short)0, c, false, false);
}

// ---------------- setup kernels ----------------

__global__ __launch_bounds__(256) void init_k(float* __restrict__ DEG,
                                              float* __restrict__ DEGR,
                                              float* __restrict__ SCAL) {
  const int i = blockIdx.x * 256 + threadIdx.x;
  if (i < N) { DEG[i] = 1.0f; DEGR[i] = 0.0f; }  // deg gets +1 self-loop
  if (i < 64) SCAL[i] = 0.0f;
}

__global__ __launch_bounds__(256) void convx_k(const float* __restrict__ x,
                                               unsigned short* __restrict__ XH) {
  const size_t i = (size_t)blockIdx.x * 256 + threadIdx.x;
  XH[i] = f2bf(x[i]);
}

// W1 [F_IN,H] f32 -> transposed bf16 [H][F_IN]
__global__ __launch_bounds__(256) void convw1_k(const float* __restrict__ W,
                                                unsigned short* __restrict__ Wt) {
  const int t = blockIdx.x * 256 + threadIdx.x;  // t = n*F_IN + k
  const int n = t >> 7, k = t & 127;
  Wt[t] = f2bf(W[(size_t)k * H + n]);
}

// W2 [H,K] f32 -> transposed, zero-padded bf16 [KP][H]
__global__ __launch_bounds__(256) void convw2_k(const float* __restrict__ W,
                                                unsigned short* __restrict__ Wt) {
  const int t = blockIdx.x * 256 + threadIdx.x;  // t = n*H + k
  const int n = t >> 8, k = t & 255;
  Wt[t] = (n < K) ? f2bf(W[(size_t)k * K + n]) : (unsigned short)0;
}

__global__ __launch_bounds__(256) void edge_deg_k(const int* __restrict__ ei,
                                                  const float* __restrict__ w,
                                                  float* __restrict__ DEG,
                                                  float* __restrict__ DEGR,
                                                  float* __restrict__ SCAL) {
  const int e = blockIdx.x * 256 + threadIdx.x;
  float v = 0.0f;
  if (e < E) {
    v = w[e];
    atomAddF(DEG + ei[E + e], v);                // segment_sum over col
    atomAddF(DEGR + ei[e], v);                   // segment_sum over row
  }
#pragma unroll
  for (int o = 16; o > 0; o >>= 1) v += __shfl_xor(v, o, 32);
  if ((threadIdx.x & 31) == 0) atomAddF(&SCAL[0], v);   // m = sum(w)
}

__global__ __launch_bounds__(256) void dis_k(const float* __restrict__ DEG,
                                             float* __restrict__ DIS) {
  const int i = blockIdx.x * 256 + threadIdx.x;
  if (i < N) DIS[i] = rsqrtf(DEG[i]);
}

__global__ __launch_bounds__(256) void norm_k(const int* __restrict__ ei,
                                              const float* __restrict__ w,
                                              const float* __restrict__ DIS,
                                              float* __restrict__ NORM) {
  const int e = blockIdx.x * 256 + threadIdx.x;
  if (e < E) NORM[e] = DIS[ei[e]] * w[e] * DIS[ei[E + e]];
}

// ---------------- layer 1 ----------------

// 16 rows x 256 cols per block; 8 waves, wave w -> cols [32w, 32w+32), conv + lin paths.
__global__ __launch_bounds__(256) void gemm1_k(const unsigned short* __restrict__ XH,
                                               const unsigned short* __restrict__ WCT,
                                               const unsigned short* __restrict__ WLT,
                                               const float* __restrict__ bc,
                                               const float* __restrict__ bl,
                                               const float* __restrict__ dis,
                                               float* __restrict__ H1,
                                               float* __restrict__ AGG) {
  const int r0 = blockIdx.x * 16;
  const int c0 = (threadIdx.x >> 5) * 32;
  v8f aC0 = {}, aC1 = {}, aL0 = {}, aL1 = {};
#pragma unroll
  for (int k0 = 0; k0 < F_IN; k0 += 32) {
    v16bf a  = load_a(XH, F_IN, r0, k0);
    v16bf b0 = load_b(WCT, F_IN, k0, c0);
    v16bf b1 = load_b(WCT, F_IN, k0, c0 + 16);
    v16bf b2 = load_b(WLT, F_IN, k0, c0);
    v16bf b3 = load_b(WLT, F_IN, k0, c0 + 16);
    aC0 = wmma_bf16(a, b0, aC0);
    aC1 = wmma_bf16(a, b1, aC1);
    aL0 = wmma_bf16(a, b2, aL0);
    aL1 = wmma_bf16(a, b3, aL1);
  }
  const int l = threadIdx.x & 31;
  const int n = l & 15;
  const int mo = (l >> 4) << 3;
#pragma unroll
  for (int j = 0; j < 8; ++j) {
    const int gr = r0 + j + mo;
    const float d  = dis[gr];
    const float d2 = d * d;
    const size_t rb = (size_t)gr * H;
    {
      const int col = c0 + n;
      const float hc = aC0[j];
      H1[rb + col]  = hc;
      AGG[rb + col] = bc[col] + bl[col] + aL0[j] + d2 * hc;
    }
    {
      const int col = c0 + 16 + n;
      const float hc = aC1[j];
      H1[rb + col]  = hc;
      AGG[rb + col] = bc[col] + bl[col] + aL1[j] + d2 * hc;
    }
  }
}

// one wave per edge: 256 features, coalesced loads + fp32 global atomics
__global__ __launch_bounds__(256) void scatter1_k(const int* __restrict__ ei,
                                                  const float* __restrict__ nrm,
                                                  const float* __restrict__ H1,
                                                  float* __restrict__ AGG) {
  const int e = blockIdx.x * 8 + (threadIdx.x >> 5);
  const int lane = threadIdx.x & 31;
  const int r = ei[e], c = ei[E + e];
  const float w = nrm[e];
  const float* __restrict__ src = H1 + (size_t)r * H;
  float* __restrict__ dst = AGG + (size_t)c * H;
#pragma unroll
  for (int i = 0; i < 8; ++i) {
    const int f = i * 32 + lane;
    atomAddF(dst + f, w * src[f]);
  }
}

__global__ __launch_bounds__(256) void relu_k(const float* __restrict__ AGG,
                                              unsigned short* __restrict__ HB) {
  const size_t i = (size_t)blockIdx.x * 256 + threadIdx.x;
  HB[i] = f2bf(fmaxf(AGG[i], 0.0f));
}

// ---------------- layer 2 ----------------

// 32 rows x 32 cols per block; 4 waves = 2 row-tiles x 2 col-tiles.
__global__ __launch_bounds__(128) void gemm2_k(const unsigned short* __restrict__ HB,
                                               const unsigned short* __restrict__ WCT,
                                               const unsigned short* __restrict__ WLT,
                                               const float* __restrict__ bc,
                                               const float* __restrict__ bl,
                                               const float* __restrict__ dis,
                                               float* __restrict__ HC2,
                                               float* __restrict__ OUTP) {
  const int wave = threadIdx.x >> 5;
  const int r0 = blockIdx.x * 32 + (wave >> 1) * 16;
  const int c0 = (wave & 1) * 16;
  v8f aC = {}, aL = {};
#pragma unroll
  for (int k0 = 0; k0 < H; k0 += 32) {
    v16bf a  = load_a(HB, H, r0, k0);
    v16bf b0 = load_b(WCT, H, k0, c0);
    v16bf b1 = load_b(WLT, H, k0, c0);
    aC = wmma_bf16(a, b0, aC);
    aL = wmma_bf16(a, b1, aL);
  }
  const int l = threadIdx.x & 31;
  const int n = l & 15;
  const int mo = (l >> 4) << 3;
#pragma unroll
  for (int j = 0; j < 8; ++j) {
    const int gr  = r0 + j + mo;
    const float d = dis[gr];
    const int col = c0 + n;
    const float hc = aC[j];
    const float bias = (col < K) ? (bc[col] + bl[col]) : 0.0f;
    HC2[(size_t)gr * KP + col]  = hc;
    OUTP[(size_t)gr * KP + col] = bias + aL[j] + d * d * hc;
  }
}

// one wave per edge; lanes 0..29 each add one cluster channel
__global__ __launch_bounds__(256) void scatter2_k(const int* __restrict__ ei,
                                                  const float* __restrict__ nrm,
                                                  const float* __restrict__ HC2,
                                                  float* __restrict__ OUTP) {
  const int e = blockIdx.x * 8 + (threadIdx.x >> 5);
  const int lane = threadIdx.x & 31;
  if (lane < K) {
    const int r = ei[e], c = ei[E + e];
    atomAddF(OUTP + (size_t)c * KP + lane, nrm[e] * HC2[(size_t)r * KP + lane]);
  }
}

// ---------------- softmax + loss ----------------

__global__ __launch_bounds__(256) void softmax_k(const float* __restrict__ OUTP,
                                                 const float* __restrict__ degr,
                                                 float* __restrict__ FX,
                                                 float* __restrict__ SCAL) {
  __shared__ float shs[K];   // sum FX*degree per cluster
  __shared__ float shc[K];   // column sums per cluster
  if (threadIdx.x < K) { shs[threadIdx.x] = 0.0f; shc[threadIdx.x] = 0.0f; }
  __syncthreads();
  const int i = blockIdx.x * 256 + threadIdx.x;
  if (i < N) {
    float v[K];
    const float* __restrict__ row = OUTP + (size_t)i * KP;
    float mx = -3.0e38f;
#pragma unroll
    for (int k = 0; k < K; ++k) { v[k] = row[k]; mx = fmaxf(mx, v[k]); }
    float s = 0.0f;
#pragma unroll
    for (int k = 0; k < K; ++k) { v[k] = __expf(v[k] - mx); s += v[k]; }
    const float inv = 1.0f / s;
    const float dg = degr[i];
#pragma unroll
    for (int k = 0; k < K; ++k) {
      const float fx = fminf(fmaxf(v[k] * inv, 1e-9f), 1.0f - 1e-9f);
      FX[(size_t)i * K + k] = fx;
      atomicAdd(&shs[k], fx * dg);   // ds_add_f32
      atomicAdd(&shc[k], fx);
    }
  }
  __syncthreads();
  if (threadIdx.x < K) {
    atomAddF(&SCAL[2 + threadIdx.x],  shs[threadIdx.x]);
    atomAddF(&SCAL[32 + threadIdx.x], shc[threadIdx.x]);
  }
}

__global__ __launch_bounds__(256) void conn_k(const int* __restrict__ ei,
                                              const float* __restrict__ w,
                                              const float* __restrict__ FX,
                                              float* __restrict__ SCAL) {
  const int e = blockIdx.x * 256 + threadIdx.x;
  float v = 0.0f;
  if (e < E) {
    const int r = ei[e], c = ei[E + e];
    const float* __restrict__ fr = FX + (size_t)r * K;
    const float* __restrict__ fc = FX + (size_t)c * K;
    float acc = 0.0f;
#pragma unroll
    for (int k = 0; k < K - 1; ++k) acc += fr[k] * fc[k];   // [:-1]
    v = w[e] * acc;
  }
#pragma unroll
  for (int o = 16; o > 0; o >>= 1) v += __shfl_xor(v, o, 32);
  if ((threadIdx.x & 31) == 0) atomAddF(&SCAL[1], v);
}

__global__ void finalize_k(const float* __restrict__ SCAL, float* __restrict__ loss_out) {
  const float m = SCAL[0];
  const float conn = SCAL[1];
  float ss = 0.0f;
  for (int k = 0; k < K - 1; ++k) ss += SCAL[2 + k] * SCAL[2 + k];
  const float avg = ss / (2.0f * m);
  const float mod = -(conn - avg) / (2.0f * m);
  float cs = 0.0f;
  for (int k = 0; k < K; ++k) cs += SCAL[32 + k] * SCAL[32 + k];
  const float reg = sqrtf(cs + 1e-9f) * sqrtf((float)K) / (float)N - 1.0f;
  *loss_out = mod + REGC * reg;
}

// ---------------- launch ----------------

extern "C" void kernel_launch(void* const* d_in, const int* in_sizes, int n_in,
                              void* d_out, int out_size, void* d_ws, size_t ws_size,
                              hipStream_t stream) {
  const float* x   = (const float*)d_in[0];
  const int*   ei  = (const int*)d_in[1];
  const float* ew  = (const float*)d_in[2];
  const float* Wc1 = (const float*)d_in[3];
  const float* bc1 = (const float*)d_in[4];
  const float* Wl1 = (const float*)d_in[5];
  const float* bl1 = (const float*)d_in[6];
  const float* Wc2 = (const float*)d_in[7];
  const float* bc2 = (const float*)d_in[8];
  const float* Wl2 = (const float*)d_in[9];
  const float* bl2 = (const float*)d_in[10];
  float* FX = (float*)d_out;                     // [N*K] + 1 scalar loss

  char* ws = (char*)d_ws;
  size_t off = 0;
  auto alloc = [&](size_t bytes) -> char* {
    char* p = ws + off;
    off = (off + bytes + 255) & ~(size_t)255;
    return p;
  };
  unsigned short* XH   = (unsigned short*)alloc((size_t)N * F_IN * 2);  // x in bf16
  unsigned short* W1CT = (unsigned short*)alloc((size_t)H * F_IN * 2);  // transposed bf16 weights
  unsigned short* W1LT = (unsigned short*)alloc((size_t)H * F_IN * 2);
  unsigned short* W2CT = (unsigned short*)alloc((size_t)KP * H * 2);
  unsigned short* W2LT = (unsigned short*)alloc((size_t)KP * H * 2);
  float* H1   = (float*)alloc((size_t)N * H * 4);   // x@W_conv1 (pre-aggregation)
  float* AGG  = (float*)alloc((size_t)N * H * 4);   // aggregation accumulator / layer1 out
  float* DEG  = (float*)alloc((size_t)N * 4);
  float* DIS  = (float*)alloc((size_t)N * 4);
  float* DEGR = (float*)alloc((size_t)N * 4);
  float* NORM = (float*)alloc((size_t)E * 4);
  float* SCAL = (float*)alloc(256);                 // [0]=m [1]=conn [2..31]=s [32..61]=colsum
  // aliases over buffers that are dead by the time these are written:
  unsigned short* HB = (unsigned short*)H1;                      // bf16 hidden (H1 dead after scatter1)
  float* HC2  = (float*)((char*)H1 + (size_t)N * H * 2);         // fits in back half of H1
  float* OUTP = AGG;                                             // AGG dead after relu

  init_k    <<<(N + 255) / 256, 256, 0, stream>>>(DEG, DEGR, SCAL);
  convx_k   <<<(N * F_IN) / 256, 256, 0, stream>>>(x, XH);
  convw1_k  <<<(F_IN * H) / 256, 256, 0, stream>>>(Wc1, W1CT);
  convw1_k  <<<(F_IN * H) / 256, 256, 0, stream>>>(Wl1, W1LT);
  convw2_k  <<<(KP * H) / 256, 256, 0, stream>>>(Wc2, W2CT);
  convw2_k  <<<(KP * H) / 256, 256, 0, stream>>>(Wl2, W2LT);
  edge_deg_k<<<E / 256, 256, 0, stream>>>(ei, ew, DEG, DEGR, SCAL);
  dis_k     <<<(N + 255) / 256, 256, 0, stream>>>(DEG, DIS);
  norm_k    <<<E / 256, 256, 0, stream>>>(ei, ew, DIS, NORM);
  gemm1_k   <<<N / 16, 256, 0, stream>>>(XH, W1CT, W1LT, bc1, bl1, DIS, H1, AGG);
  scatter1_k<<<E / 8, 256, 0, stream>>>(ei, NORM, H1, AGG);
  relu_k    <<<(N * H) / 256, 256, 0, stream>>>(AGG, HB);
  gemm2_k   <<<N / 32, 128, 0, stream>>>(HB, W2CT, W2LT, bc2, bl2, DIS, HC2, OUTP);
  scatter2_k<<<E / 8, 256, 0, stream>>>(ei, NORM, HC2, OUTP);
  softmax_k <<<(N + 255) / 256, 256, 0, stream>>>(OUTP, DEGR, FX, SCAL);
  conn_k    <<<E / 256, 256, 0, stream>>>(ei, ew, FX, SCAL);
  finalize_k<<<1, 1, 0, stream>>>(SCAL, FX + (size_t)N * K);
}
